// SelfAttention_11828339933196
// MI455X (gfx1250) — matrix operand order repaired and checked by reading, not compiled, not run on hardware
//
#include <hip/hip_runtime.h>

typedef __attribute__((ext_vector_type(16))) _Float16 v16h;
typedef __attribute__((ext_vector_type(4)))  _Float16 h4;
typedef __attribute__((ext_vector_type(2)))  _Float16 h2;
typedef __attribute__((ext_vector_type(2)))  __fp16   pkh2;
typedef __attribute__((ext_vector_type(8)))  float    v8f;
typedef __attribute__((ext_vector_type(4)))  float    f32x4;

#define SEQ 8192
#define DIM 128

constexpr int KV_CHUNK      = 64;              // KV rows per main-loop step
constexpr int WAVES         = 4;
constexpr int BLOCK_THREADS = WAVES * 32;      // 128
constexpr int QROWS_PER_WAVE  = 16;
constexpr int QROWS_PER_BLOCK = WAVES * QROWS_PER_WAVE;  // 64

__device__ __forceinline__ v8f wmma_f16(v16h a, v16h b, v8f c) {
  return __builtin_amdgcn_wmma_f32_16x16x32_f16(
      /*neg_a=*/false, a, /*neg_b=*/false, b,
      /*c_mod=*/(short)0, c, /*reuse_a=*/false, /*reuse_b=*/false);
}

__device__ __forceinline__ h2 cvt_pk_f16(float a, float b) {
  pkh2 t = __builtin_amdgcn_cvt_pkrtz(a, b);   // v_cvt_pk_f16_f32
  return __builtin_bit_cast(h2, t);
}

__device__ __forceinline__ h4 join4(h2 lo, h2 hi) {
  return __builtin_shufflevector(lo, hi, 0, 1, 2, 3);
}

__global__ __launch_bounds__(BLOCK_THREADS)
void flash_attn_f16_wmma(const float* __restrict__ Q,
                         const float* __restrict__ K,
                         const float* __restrict__ V,
                         float* __restrict__ Out) {
  // LDS staging: K chunk row-major [kv][d], V chunk transposed [d][kv],
  // and a per-wave P buffer with column-interleaved layout:
  //   P[row][col] lives at sP[wave][row*64 + 4*(col&15) + (col>>4)]
  __shared__ __align__(32) _Float16 sK [KV_CHUNK * DIM];   // 16 KB
  __shared__ __align__(32) _Float16 sVT[DIM * KV_CHUNK];   // 16 KB
  __shared__ __align__(32) _Float16 sP [WAVES][QROWS_PER_WAVE * KV_CHUNK]; // 8 KB

  const int tid  = threadIdx.x;
  const int wave = tid >> 5;
  const int lane = tid & 31;
  const int hi   = lane >> 4;     // which 16-lane half
  const int l16  = lane & 15;

  const int qbase = blockIdx.x * QROWS_PER_BLOCK + wave * QROWS_PER_WAVE;

  // Fold softmax scale AND log2(e) into Q so the QK^T WMMA output is already
  // in exp2 domain: S' = (Q.K^T)/sqrt(D)*log2e. Softmax is base-invariant.
  const float qscale = 0.08838834764831845f * 1.4426950408889634f;

  // ---- Load Q tile once as four 16x32 f16 A-fragments.
  // A order used here (16x32, contiguous k, matches B): row M = lane&15; slots 0..15 hold K = hi*16 + j.
  v16h qf[4];
  {
    const float* qrow = Q + (qbase + l16) * DIM;
    const int kb = hi * 16;
#pragma unroll
    for (int c = 0; c < 4; ++c) {
#pragma unroll
      for (int j = 0; j < 8; ++j) {
        qf[c][j]     = (_Float16)(qrow[c * 32 + kb + j]      * qscale);
        qf[c][8 + j] = (_Float16)(qrow[c * 32 + 8 + kb + j] * qscale);
      }
    }
  }

  // ---- Accumulators: O is 16x128 as 8 C-layout tiles; per-lane row stats.
  // C layout: lane owns rows {r + 8*hi}, column N = lane&15 of each 16-wide tile.
  v8f o[8];
#pragma unroll
  for (int t = 0; t < 8; ++t)
#pragma unroll
    for (int r = 0; r < 8; ++r) o[t][r] = 0.0f;

  v8f m, l;
#pragma unroll
  for (int r = 0; r < 8; ++r) { m[r] = -1e30f; l[r] = 0.0f; }

  for (int jkv = 0; jkv < SEQ; jkv += KV_CHUNK) {
    __syncthreads();  // protect LDS from previous iteration's readers

    // ---- Stage K (row-major, packed f16) to LDS: 16 float4 per thread.
#pragma unroll
    for (int i = 0; i < 16; ++i) {
      const int e4  = i * BLOCK_THREADS + tid;   // float4 index 0..2047
      const int row = e4 >> 5;                   // 32 float4 per 128-float row
      const int d0  = (e4 & 31) * 4;
      const f32x4 k4 = *(const f32x4*)(K + (jkv + row) * DIM + d0);
      *(h4*)&sK[row * DIM + d0] =
          join4(cvt_pk_f16(k4[0], k4[1]), cvt_pk_f16(k4[2], k4[3]));
    }

    // ---- Stage V transposed via 4x4 register-block transpose: 4 tiles/thread.
#pragma unroll
    for (int i = 0; i < 4; ++i) {
      const int tile = i * BLOCK_THREADS + tid;  // 0..511
      const int rg   = tile >> 5;                // row group: rows 4rg..4rg+3
      const int dg   = tile & 31;                // dim group: dims 4dg..4dg+3
      const float* vp = V + (jkv + 4 * rg) * DIM + 4 * dg;
      const f32x4 r0 = *(const f32x4*)(vp);
      const f32x4 r1 = *(const f32x4*)(vp + DIM);
      const f32x4 r2 = *(const f32x4*)(vp + 2 * DIM);
      const f32x4 r3 = *(const f32x4*)(vp + 3 * DIM);
#pragma unroll
      for (int c = 0; c < 4; ++c) {
        *(h4*)&sVT[(4 * dg + c) * KV_CHUNK + 4 * rg] =
            join4(cvt_pk_f16(r0[c], r1[c]), cvt_pk_f16(r2[c], r3[c]));
      }
    }
    __syncthreads();

    // ---- Prefetch next chunk's K/V lines into cache while we compute.
    if (jkv + KV_CHUNK < SEQ) {
      const char* kn = (const char*)(K + (jkv + KV_CHUNK) * DIM);
      const char* vn = (const char*)(V + (jkv + KV_CHUNK) * DIM);
#pragma unroll
      for (int p = 0; p < 2; ++p) {   // 256 x 128B lines per matrix, 128 threads
        __builtin_prefetch(kn + (p * BLOCK_THREADS + tid) * 128, 0, 1);
        __builtin_prefetch(vn + (p * BLOCK_THREADS + tid) * 128, 0, 1);
      }
    }

    // ---- S' = (Q' * K^T): four 16x16 tiles (kv col blocks of 16).
    // B order used here (32x16, contiguous k, matches A): column N = lane&15; slots 0..15 hold K = hi*16 + t.
    v8f s[4];
#pragma unroll
    for (int t = 0; t < 4; ++t)
#pragma unroll
      for (int r = 0; r < 8; ++r) s[t][r] = 0.0f;
#pragma unroll
    for (int c = 0; c < 4; ++c) {
      const _Float16* kbase = &sK[l16 * DIM + c * 32 + hi * 16];
      const v16h b0 = *(const v16h*)(kbase);
      const v16h b1 = *(const v16h*)(kbase + 16 * DIM);
      const v16h b2 = *(const v16h*)(kbase + 32 * DIM);
      const v16h b3 = *(const v16h*)(kbase + 48 * DIM);
      s[0] = wmma_f16(qf[c], b0, s[0]);
      s[1] = wmma_f16(qf[c], b1, s[1]);
      s[2] = wmma_f16(qf[c], b2, s[2]);
      s[3] = wmma_f16(qf[c], b3, s[3]);
    }

    // ---- Online softmax update (exp2 domain).
    v8f mnew;
#pragma unroll
    for (int r = 0; r < 8; ++r) {
      float mx = fmaxf(fmaxf(s[0][r], s[1][r]), fmaxf(s[2][r], s[3][r]));
      mx = fmaxf(mx, __shfl_xor(mx, 1));
      mx = fmaxf(mx, __shfl_xor(mx, 2));
      mx = fmaxf(mx, __shfl_xor(mx, 4));
      mx = fmaxf(mx, __shfl_xor(mx, 8));
      mnew[r] = fmaxf(m[r], mx);
    }
    v8f av;                       // alpha = exp2(m_old - m_new)
    const v8f ad = m - mnew;
#pragma unroll
    for (int r = 0; r < 8; ++r) av[r] = __builtin_amdgcn_exp2f(ad[r]);
    m = mnew;

    v8f p[4];
#pragma unroll
    for (int t = 0; t < 4; ++t) {
      const v8f d = s[t] - mnew;  // vector sub
#pragma unroll
      for (int r = 0; r < 8; ++r) p[t][r] = __builtin_amdgcn_exp2f(d[r]);
    }
    l = l * av + ((p[0] + p[1]) + (p[2] + p[3]));

    // ---- Write P to LDS (column-interleaved): one b64 store per row.
#pragma unroll
    for (int r = 0; r < 8; ++r) {
      *(h4*)&sP[wave][(r + hi * 8) * KV_CHUNK + 4 * l16] =
          join4(cvt_pk_f16(p[0][r], p[1][r]), cvt_pk_f16(p[2][r], p[3][r]));
    }

    // ---- Re-load P as two 16x32 A-fragments (kv 0..31 and 32..63).
    // Interleaved layout: pa0[s] (kv 16*hi+s) at [4*s + hi], pa1[s] at [4*s + 2 + hi]
    v16h pa0, pa1;
    {
      const _Float16* pr = &sP[wave][l16 * KV_CHUNK];
#pragma unroll
      for (int j = 0; j < 8; ++j) {
        pa0[j]     = pr[4 * j + hi];
        pa0[8 + j] = pr[4 * (8 + j) + hi];
        pa1[j]     = pr[4 * j + 2 + hi];
        pa1[8 + j] = pr[4 * (8 + j) + 2 + hi];
      }
    }

    // ---- Rescale O by alpha (vector ops -> packed f32 muls).
#pragma unroll
    for (int t = 0; t < 8; ++t) o[t] = o[t] * av;

    // ---- O += P * V_chunk: 8 output N-tiles, two WMMAs each (K=64),
    // grouped in pairs so 8 ds_load_b128 are in flight per 4 WMMAs.
#pragma unroll
    for (int g = 0; g < 4; ++g) {
      const int t0 = 2 * g, t1 = 2 * g + 1;
      const _Float16* vb0 = &sVT[(t0 * 16 + l16) * KV_CHUNK + hi * 16];
      const _Float16* vb1 = &sVT[(t1 * 16 + l16) * KV_CHUNK + hi * 16];
      const v16h b00 = *(const v16h*)(vb0);
      const v16h b01 = *(const v16h*)(vb0 + 32);
      const v16h b10 = *(const v16h*)(vb1);
      const v16h b11 = *(const v16h*)(vb1 + 32);
      o[t0] = wmma_f16(pa0, b00, o[t0]);
      o[t0] = wmma_f16(pa1, b01, o[t0]);
      o[t1] = wmma_f16(pa0, b10, o[t1]);
      o[t1] = wmma_f16(pa1, b11, o[t1]);
    }
  }

  // ---- Epilogue: reduce row sums across the half, normalize, store fp32.
#pragma unroll
  for (int r = 0; r < 8; ++r) {
    float lt = l[r];
    lt += __shfl_xor(lt, 1);
    lt += __shfl_xor(lt, 2);
    lt += __shfl_xor(lt, 4);
    lt += __shfl_xor(lt, 8);
    const float inv = 1.0f / lt;
    const int row = qbase + r + hi * 8;
#pragma unroll
    for (int t = 0; t < 8; ++t) {
      Out[row * DIM + t * 16 + l16] = o[t][r] * inv;
    }
  }
}

extern "C" void kernel_launch(void* const* d_in, const int* in_sizes, int n_in,
                              void* d_out, int out_size, void* d_ws, size_t ws_size,
                              hipStream_t stream) {
  (void)in_sizes; (void)n_in; (void)out_size; (void)d_ws; (void)ws_size;
  const float* Q = (const float*)d_in[0];
  const float* K = (const float*)d_in[1];
  const float* V = (const float*)d_in[2];
  float* Out = (float*)d_out;

  dim3 grid(SEQ / QROWS_PER_BLOCK);   // 128 blocks
  dim3 block(BLOCK_THREADS);          // 4 waves of 32
  flash_attn_f16_wmma<<<grid, block, 0, stream>>>(Q, K, V, Out);
}
